// Align_Feature_and_Aggregation_45389214384886
// MI455X (gfx1250) — matrix operand order repaired and checked by reading, not compiled, not run on hardware
//
#include <hip/hip_runtime.h>

typedef __attribute__((ext_vector_type(16))) _Float16 v16h;
typedef __attribute__((ext_vector_type(8)))  float    v8f;

constexpr int Cc   = 128;     // channels
constexpr int Hh   = 200;
constexpr int Ww   = 200;
constexpr int HW   = Hh * Ww; // 40000
constexpr int Rr   = 4;       // NEI//2
constexpr int NOFF = 81;      // 9x9 offsets

// ---------------------------------------------------------------------------
// 1x1 conv + bias + ReLU as WMMA GEMM, register-blocked 2x2 tiles per wave:
// each wave computes 32 out-channels x 32 pixels. Per k-step: 2 A-frag loads +
// 2 B-frag loads feed 4 WMMAs (16 loads/WMMA instead of 32).
// Logical 8-image batch: imgs 0..3 from in0, 4..7 from in1 (each (4,CIN,HW)).
// w: (COUT,CIN), out: (8,COUT,HW).
// ---------------------------------------------------------------------------
__global__ __launch_bounds__(32)
void conv1x1_relu_wmma(const float* __restrict__ in0, const float* __restrict__ in1,
                       const float* __restrict__ wgt, const float* __restrict__ bias,
                       float* __restrict__ out, int CIN, int COUT)
{
    const int  lane   = threadIdx.x;
    const bool hi     = lane >= 16;
    const int  l      = lane & 15;
    const int  p_base = blockIdx.x * 32;    // 2 pixel tiles
    const int  m_base = blockIdx.y * 32;    // 2 out-channel tiles
    const int  img    = blockIdx.z;
    const float* xin  = (img < 4) ? (in0 + (size_t)img * CIN * HW)
                                  : (in1 + (size_t)(img - 4) * CIN * HW);
    float* yout = out + (size_t)img * COUT * HW;

    v8f acc[2][2];
#pragma unroll
    for (int mt = 0; mt < 2; ++mt)
#pragma unroll
        for (int nt = 0; nt < 2; ++nt) acc[mt][nt] = {};

    for (int kb = 0; kb < CIN; kb += 32) {
        v16h am[2], bm[2];
#pragma unroll
        for (int nt = 0; nt < 2; ++nt) {        // B: (K x N) input pixels
            const int p = p_base + nt * 16 + l;
#pragma unroll
            for (int i = 0; i < 16; ++i) {
                int kB = kb + (hi ? 16 : 0) + i;
                bm[nt][i] = (_Float16)xin[(size_t)kB * HW + p];
            }
        }
#pragma unroll
        for (int mt = 0; mt < 2; ++mt) {        // A: (M x K) weights
            const int m = m_base + mt * 16 + l;
#pragma unroll
            for (int i = 0; i < 16; ++i) {
                int kA = kb + (hi ? 16 : 0) + i;
                am[mt][i] = (_Float16)wgt[(size_t)m * CIN + kA];
            }
        }
#pragma unroll
        for (int mt = 0; mt < 2; ++mt)
#pragma unroll
            for (int nt = 0; nt < 2; ++nt)
                acc[mt][nt] = __builtin_amdgcn_wmma_f32_16x16x32_f16(
                    false, am[mt], false, bm[nt], (short)0, acc[mt][nt], false, false);
    }

#pragma unroll
    for (int mt = 0; mt < 2; ++mt) {
#pragma unroll
        for (int nt = 0; nt < 2; ++nt) {
            const int p = p_base + nt * 16 + l;  // D: M=r(+8), N=lane&15
#pragma unroll
            for (int r = 0; r < 8; ++r) {
                int m = m_base + mt * 16 + r + (hi ? 8 : 0);
                float v = acc[mt][nt][r] + bias[m];
                yout[(size_t)m * HW + p] = fmaxf(v, 0.f);
            }
        }
    }
}

// ---------------------------------------------------------------------------
// 3x3 conv (pad=1) + bias + ReLU as WMMA GEMM (im2col on the fly, K=CIN*9),
// register-blocked 2 pixel tiles x 2 out-channel tiles per wave.
// in: (8,CIN,HW), w: (COUT,CIN,3,3), out: (8,COUT,HW).
// ---------------------------------------------------------------------------
__global__ __launch_bounds__(32)
void conv3x3_relu_wmma(const float* __restrict__ in, const float* __restrict__ wgt,
                       const float* __restrict__ bias, float* __restrict__ out,
                       int CIN, int COUT)
{
    const int  lane   = threadIdx.x;
    const bool hi     = lane >= 16;
    const int  l      = lane & 15;
    const int  p_base = blockIdx.x * 32;    // 2 pixel tiles
    const int  m_base = blockIdx.y * 32;    // 2 out-channel tiles
    const int  img    = blockIdx.z;
    const float* xin  = in  + (size_t)img * CIN * HW;
    float* yout       = out + (size_t)img * COUT * HW;

    v8f acc[2][2];
#pragma unroll
    for (int mt = 0; mt < 2; ++mt)
#pragma unroll
        for (int nt = 0; nt < 2; ++nt) acc[mt][nt] = {};

#pragma unroll
    for (int tap = 0; tap < 9; ++tap) {
        const int di = tap / 3 - 1, dj = tap % 3 - 1;
        // neighbor pixel (and validity) for each pixel tile, this lane's column
        int  q[2];
        bool ok[2];
#pragma unroll
        for (int nt = 0; nt < 2; ++nt) {
            const int p   = p_base + nt * 16 + l;
            const int row = p / Ww, col = p % Ww;
            const int rr = row + di, cc = col + dj;
            ok[nt] = (rr >= 0) && (rr < Hh) && (cc >= 0) && (cc < Ww);
            q[nt]  = ok[nt] ? (rr * Ww + cc) : 0;
        }
        for (int kb = 0; kb < CIN; kb += 32) {
            v16h am[2], bm[2];
#pragma unroll
            for (int nt = 0; nt < 2; ++nt) {
#pragma unroll
                for (int i = 0; i < 16; ++i) {
                    int kB = kb + (hi ? 16 : 0) + i;
                    float v = ok[nt] ? xin[(size_t)kB * HW + q[nt]] : 0.f;
                    bm[nt][i] = (_Float16)v;
                }
            }
#pragma unroll
            for (int mt = 0; mt < 2; ++mt) {
                const int m = m_base + mt * 16 + l;
#pragma unroll
                for (int i = 0; i < 16; ++i) {
                    int kA = kb + (hi ? 16 : 0) + i;
                    am[mt][i] = (_Float16)wgt[((size_t)m * CIN + kA) * 9 + tap];
                }
            }
#pragma unroll
            for (int mt = 0; mt < 2; ++mt)
#pragma unroll
                for (int nt = 0; nt < 2; ++nt)
                    acc[mt][nt] = __builtin_amdgcn_wmma_f32_16x16x32_f16(
                        false, am[mt], false, bm[nt], (short)0, acc[mt][nt], false, false);
        }
    }

#pragma unroll
    for (int mt = 0; mt < 2; ++mt) {
#pragma unroll
        for (int nt = 0; nt < 2; ++nt) {
            const int p = p_base + nt * 16 + l;
#pragma unroll
            for (int r = 0; r < 8; ++r) {
                int m = m_base + mt * 16 + r + (hi ? 8 : 0);
                float v = acc[mt][nt][r] + bias[m];
                yout[(size_t)m * HW + p] = fmaxf(v, 0.f);
            }
        }
    }
}

// ---------------------------------------------------------------------------
// Same kernel but with a single 16-wide out-channel tile (for COUT=32 with
// 2 pixel tiles it is covered by gridDim.y=1 of the blocked kernel; ag2 uses
// COUT=32 which is exactly 2 m-tiles, so the blocked kernel handles it too).
// ---------------------------------------------------------------------------

// ---------------------------------------------------------------------------
// Local cross-correlation: corr[img,k,p] = sum_c e_sel[c,p] * e_cur[c, p+off_k]
// e: (8,64,HW) (imgs 0..3 = e_sel, 4..7 = e_cur). Zero outside image (pad).
// ---------------------------------------------------------------------------
__global__ __launch_bounds__(256)
void corr_kernel(const float* __restrict__ e, float* __restrict__ corr)
{
    const int p = blockIdx.x * 256 + threadIdx.x;
    if (p >= HW) return;
    const int img = blockIdx.y;
    const float* es = e + (size_t)img * 64 * HW;
    const float* ec = e + (size_t)(img + 4) * 64 * HW;
    const int row = p / Ww, col = p % Ww;

    float sv[64];
#pragma unroll
    for (int c = 0; c < 64; ++c) sv[c] = es[(size_t)c * HW + p];

    int kidx = 0;
    for (int di = -Rr; di <= Rr; ++di) {
        for (int dj = -Rr; dj <= Rr; ++dj) {
            const int rr = row + di, cc = col + dj;
            float acc = 0.f;
            if (rr >= 0 && rr < Hh && cc >= 0 && cc < Ww) {
                const int q = rr * Ww + cc;
#pragma unroll
                for (int c = 0; c < 64; ++c) acc += sv[c] * ec[(size_t)c * HW + q];
            }
            corr[((size_t)img * NOFF + kidx) * HW + p] = acc;
            ++kidx;
        }
    }
}

// ---------------------------------------------------------------------------
// Softmax over the 81 offset channels, in place. corr: (4,81,HW).
// ---------------------------------------------------------------------------
__global__ __launch_bounds__(256)
void softmax81_kernel(float* __restrict__ corr)
{
    const int p = blockIdx.x * 256 + threadIdx.x;
    if (p >= HW) return;
    const int img = blockIdx.y;
    float* base = corr + (size_t)img * NOFF * HW + p;
    float m = -3.4e38f;
    for (int k = 0; k < NOFF; ++k) m = fmaxf(m, base[(size_t)k * HW]);
    float s = 0.f;
    for (int k = 0; k < NOFF; ++k) s += __expf(base[(size_t)k * HW] - m);
    const float inv = 1.f / s;
    for (int k = 0; k < NOFF; ++k)
        base[(size_t)k * HW] = __expf(base[(size_t)k * HW] - m) * inv;
}

// ---------------------------------------------------------------------------
// align[img,c,p] = sum_k w[img,k,p] * fs[img,c, p+off_k]  (zero pad outside).
// ---------------------------------------------------------------------------
__global__ __launch_bounds__(256)
void align_kernel(const float* __restrict__ wgt, const float* __restrict__ fs,
                  float* __restrict__ algn)
{
    const int p = blockIdx.x * 256 + threadIdx.x;
    if (p >= HW) return;
    const int img = blockIdx.y;
    const int row = p / Ww, col = p % Ww;

    float wk[NOFF];
    int   qk[NOFF];
#pragma unroll
    for (int di = -Rr; di <= Rr; ++di) {
#pragma unroll
        for (int dj = -Rr; dj <= Rr; ++dj) {
            const int k  = (di + Rr) * 9 + (dj + Rr);
            const int rr = row + di, cc = col + dj;
            const bool ok = (rr >= 0) && (rr < Hh) && (cc >= 0) && (cc < Ww);
            wk[k] = ok ? wgt[((size_t)img * NOFF + k) * HW + p] : 0.f;
            qk[k] = ok ? (rr * Ww + cc) : p;
        }
    }
    const float* fsb = fs   + (size_t)img * Cc * HW;
    float*       ab  = algn + (size_t)img * Cc * HW;
    for (int c = 0; c < Cc; ++c) {
        const float* fsc = fsb + (size_t)c * HW;
        float acc = 0.f;
#pragma unroll
        for (int k = 0; k < NOFF; ++k) acc += wk[k] * fsc[qk[k]];
        ab[(size_t)c * HW + p] = acc;
    }
}

// ---------------------------------------------------------------------------
// Fused ag3 (1x1 32->1 + ReLU) on both halves + 2-way softmax + final blend.
// a2: (8,32,HW), out = w0*align + w1*feature_current, (4,128,HW).
// ---------------------------------------------------------------------------
__global__ __launch_bounds__(256)
void final_kernel(const float* __restrict__ a2, const float* __restrict__ w3,
                  const float* __restrict__ b3, const float* __restrict__ algn,
                  const float* __restrict__ fcur, float* __restrict__ out)
{
    const int p = blockIdx.x * 256 + threadIdx.x;
    if (p >= HW) return;
    const int img = blockIdx.y;
    float sa = b3[0], sc = b3[0];
#pragma unroll
    for (int c = 0; c < 32; ++c) {
        const float w = w3[c];
        sa += w * a2[((size_t)img * 32 + c) * HW + p];
        sc += w * a2[((size_t)(img + 4) * 32 + c) * HW + p];
    }
    sa = fmaxf(sa, 0.f);
    sc = fmaxf(sc, 0.f);
    const float m  = fmaxf(sa, sc);
    const float e0 = __expf(sa - m), e1 = __expf(sc - m);
    const float inv = 1.f / (e0 + e1);
    const float w0 = e0 * inv, w1 = e1 * inv;
    for (int c = 0; c < Cc; ++c) {
        const size_t idx = ((size_t)img * Cc + c) * HW + p;
        out[idx] = w0 * algn[idx] + w1 * fcur[idx];
    }
}

// ---------------------------------------------------------------------------
extern "C" void kernel_launch(void* const* d_in, const int* in_sizes, int n_in,
                              void* d_out, int out_size, void* d_ws, size_t ws_size,
                              hipStream_t stream)
{
    (void)in_sizes; (void)n_in; (void)out_size; (void)ws_size;
    const float* fs   = (const float*)d_in[0];
    const float* fc   = (const float*)d_in[1];
    const float* ec1w = (const float*)d_in[2];
    const float* ec1b = (const float*)d_in[3];
    const float* ec2w = (const float*)d_in[4];
    const float* ec2b = (const float*)d_in[5];
    const float* ag1w = (const float*)d_in[6];
    const float* ag1b = (const float*)d_in[7];
    const float* ag2w = (const float*)d_in[8];
    const float* ag2b = (const float*)d_in[9];
    const float* ag3w = (const float*)d_in[10];
    const float* ag3b = (const float*)d_in[11];
    float* out = (float*)d_out;

    // Workspace layout (floats): A/B reused across the two conv stages.
    float* bufA = (float*)d_ws;                  // 8*64*HW  : e1, later ag1 out
    float* bufB = bufA + (size_t)8 * 64 * HW;    // 8*64*HW  : e2, later ag2 out
    float* bufC = bufB + (size_t)8 * 64 * HW;    // 4*81*HW  : corr -> weights
    float* bufD = bufC + (size_t)4 * NOFF * HW;  // 4*128*HW : align

    const dim3 gC64 (HW / 32, 64 / 32, 8);   // 32x32 tiles, 64 out ch, 8 imgs
    const dim3 gC32 (HW / 32, 32 / 32, 8);   // 32x32 tiles, 32 out ch
    const dim3 gPix ((HW + 255) / 256, 4);

    // ec1: 1x1 128->64 + ReLU on concat(feature_select, feature_current)
    conv1x1_relu_wmma<<<gC64, 32, 0, stream>>>(fs, fc, ec1w, ec1b, bufA, Cc, 64);
    // ec2: 3x3 64->64 + ReLU
    conv3x3_relu_wmma<<<gC64, 32, 0, stream>>>(bufA, ec2w, ec2b, bufB, 64, 64);
    // 81-offset correlation + softmax
    corr_kernel     <<<gPix, 256, 0, stream>>>(bufB, bufC);
    softmax81_kernel<<<gPix, 256, 0, stream>>>(bufC);
    // weighted alignment of feature_select
    align_kernel    <<<gPix, 256, 0, stream>>>(bufC, fs, bufD);
    // ag1: 1x1 128->64 + ReLU on concat(align, feature_current)
    conv1x1_relu_wmma<<<gC64, 32, 0, stream>>>(bufD, fc, ag1w, ag1b, bufA, Cc, 64);
    // ag2: 3x3 64->32 + ReLU
    conv3x3_relu_wmma<<<gC32, 32, 0, stream>>>(bufA, ag2w, ag2b, bufB, 64, 32);
    // ag3 + 2-way softmax + blend
    final_kernel    <<<gPix, 256, 0, stream>>>(bufB, ag3w, ag3b, bufD, fc, out);
}
